// Unidir_LSTM_18915035971916
// MI455X (gfx1250) — matrix operand order repaired and checked by reading, not compiled, not run on hardware
//
#include <hip/hip_runtime.h>
#include <math.h>

// Problem dims
#define B_DIM 256
#define T_DIM 128
#define V_DIM 32000
#define E_DIM 300
#define E_PAD 320      // pad K to multiple of 32 for 16x16x32 WMMA
#define H_DIM 512
#define G_DIM 1536     // compacted gates: i,g,o (forget gate is dead in the reference)

typedef __attribute__((ext_vector_type(16))) __bf16 v16bf;
typedef __attribute__((ext_vector_type(8)))  float  v8f;
typedef unsigned int u32x4 __attribute__((ext_vector_type(4)));
typedef int          i32x8 __attribute__((ext_vector_type(8)));
typedef int          i32x4 __attribute__((ext_vector_type(4)));

union Frag { uint4 q[2]; v16bf v; };

__device__ __forceinline__ unsigned short f2bf(float f) {
  unsigned int u = __float_as_uint(f);
  u += 0x7FFFu + ((u >> 16) & 1u);          // round-to-nearest-even
  return (unsigned short)(u >> 16);
}
__device__ __forceinline__ float bf2f(unsigned short s) {
  return __uint_as_float(((unsigned int)s) << 16);
}
__device__ __forceinline__ float sigmoidf_(float x) { return 1.0f / (1.0f + __expf(-x)); }

// gate row remap: compacted row r (0..1535) -> original 4H row (i:0-511, g:1024-1535, o:1536-2047)
__device__ __forceinline__ int gmap(int r) { return (r < H_DIM) ? r : r + H_DIM; }

// ---- TDM: DMA a 512-row x 32-col (bf16) slice of compacted W_hh into LDS ------------
// D# per CDNA5 ISA; 6-arg builtin on this toolchain (clang-23).
// 2D tensor: data_size=2B, tensor_dim0=512, tensor_dim1=512, tile 32x512, dim0 stride 512.
// LDS result: 512 rows x 64B packed contiguously at lds_off.
__device__ __forceinline__ void tdm_load_whh(unsigned lds_off, unsigned long long gaddr) {
  u32x4 g0;
  g0[0] = 1u;                                              // count=1 (valid descriptor)
  g0[1] = lds_off;                                         // LDS byte address
  g0[2] = (unsigned)gaddr;                                 // global_addr[31:0]
  g0[3] = ((unsigned)(gaddr >> 32) & 0x01FFFFFFu) | (2u << 30); // addr[56:32] | type=2
  i32x8 g1;
  g1[0] = 0x00010000;                                      // wg_mask=0, data_size=1 (2 bytes)
  g1[1] = (int)(512u << 16);                               // tensor_dim0[15:0] in [63:48]
  g1[2] = (int)(512u << 16);                               // tensor_dim1[15:0] in [95:80]
  g1[3] = (int)(32u << 16);                                // tile_dim0=32 in [127:112]
  g1[4] = 512;                                             // tile_dim1=512 in [143:128]
  g1[5] = 512;                                             // tensor_dim0_stride[31:0]
  g1[6] = 0;
  g1[7] = 0;
  i32x4 z4 = {0, 0, 0, 0};                                 // groups 2/3: unused (2D tensor)
  i32x8 z8 = {0, 0, 0, 0, 0, 0, 0, 0};                     // extra descriptor words: unused
  __builtin_amdgcn_tensor_load_to_lds(g0, g1, z4, z4, z8, 0);
}

// ---------------- Kernel 0: weight/bias conversion + gate compaction ----------------
__global__ void lstm_prep(const float* __restrict__ Wih, const float* __restrict__ Whh,
                          const float* __restrict__ bih, const float* __restrict__ bhh,
                          unsigned short* __restrict__ Wihb, unsigned short* __restrict__ Whhb,
                          float* __restrict__ biasb) {
  int i = blockIdx.x * blockDim.x + threadIdx.x;
  const int n1 = G_DIM * E_PAD;
  const int n2 = G_DIM * H_DIM;
  if (i < n1) {
    int r = i / E_PAD, k = i - r * E_PAD;
    Wihb[i] = (k < E_DIM) ? f2bf(Wih[gmap(r) * E_DIM + k]) : (unsigned short)0;
  } else if (i < n1 + n2) {
    int j = i - n1;
    int r = j / H_DIM, k = j - r * H_DIM;
    Whhb[j] = f2bf(Whh[gmap(r) * H_DIM + k]);
  } else if (i < n1 + n2 + G_DIM) {
    int r = i - n1 - n2;
    biasb[r] = bih[gmap(r)] + bhh[gmap(r)];
  }
}

// ---------------- Kernel 1: batched input projection Gx = embed(X) @ Wih_sel^T + bias ----
__global__ __launch_bounds__(256)
void lstm_xproj(const long long* __restrict__ inp, const float* __restrict__ emb,
                const unsigned short* __restrict__ Wihb, const float* __restrict__ biasb,
                unsigned short* __restrict__ Gx) {
  __shared__ __align__(16) unsigned short xs[16][E_PAD + 8];   // 328 ushort row stride
  const int mtile = blockIdx.x;
  const int tid = threadIdx.x;

  for (int e = tid; e < 16 * E_PAD; e += 256) {
    int row = e / E_PAD, k = e - row * E_PAD;
    long long idx = inp[(size_t)mtile * 16 + row];
    if (idx > (long long)V_DIM || idx < 0) idx = 0;   // faithful to reference clamp
    float v = (k < E_DIM) ? emb[(size_t)idx * E_DIM + k] : 0.0f;
    xs[row][k] = f2bf(v);
  }
  __syncthreads();

  const int w = tid >> 5, lane = tid & 31;
  const int c = lane & 15, half = lane >> 4;

  v8f acc[12] = {};
  for (int kt = 0; kt < E_PAD / 32; ++kt) {
    Frag a;
    const unsigned short* ap = &xs[c][kt * 32 + half * 8];
    a.q[0] = *(const uint4*)ap;
    a.q[1] = *(const uint4*)(ap + 16);
#pragma unroll
    for (int j = 0; j < 12; ++j) {
      int n = (w * 12 + j) * 16 + c;
      const unsigned short* bp = &Wihb[(size_t)n * E_PAD + kt * 32 + half * 8];
      Frag b;
      b.q[0] = *(const uint4*)bp;
      b.q[1] = *(const uint4*)(bp + 16);
      acc[j] = __builtin_amdgcn_wmma_f32_16x16x32_bf16(false, a.v, false, b.v,
                                                       (short)0, acc[j], false, false);
    }
  }
#pragma unroll
  for (int j = 0; j < 12; ++j) {
    int colb = (w * 12 + j) * 16;
    float bv = biasb[colb + c];
#pragma unroll
    for (int v = 0; v < 8; ++v) {
      int m = half * 8 + v;
      size_t r = (size_t)mtile * 16 + m;
      Gx[r * G_DIM + colb + c] = f2bf(acc[j][v] + bv);
    }
  }
}

// ---------------- Kernel 2: persistent recurrence with TDM-staged weights ------------
// Block owns 16 batch rows; h double-buffered in LDS (bf16); no inter-WG sync.
// W_hh streamed via Tensor Data Mover into a 2x32KB LDS double buffer at (kt,gate)
// granularity: 48 stages/step, DMA of stage s+1 overlaps WMMA of stage s.
__global__ __launch_bounds__(256)
void lstm_recur(const unsigned short* __restrict__ Whhb, const unsigned short* __restrict__ Gx,
                float* __restrict__ out) {
  __shared__ __align__(16) unsigned short hb[2][16][520];        // 33,280 B (skewed rows)
  __shared__ __align__(16) unsigned char  stage[2][512 * 64];    // 65,536 B weight stage
  const int rb = blockIdx.x * 16;
  const int tid = threadIdx.x;
  for (int e = tid; e < 2 * 16 * 520; e += 256) ((unsigned short*)hb)[e] = 0;

  const int w = tid >> 5, lane = tid & 31;
  const int c = lane & 15, half = lane >> 4;

  // LDS byte offsets (flat shared addr has the LDS offset in addr[31:0])
  const unsigned stage_off0 = (unsigned)(unsigned long long)(uintptr_t)&stage[0][0];
  const unsigned long long whh_base = (unsigned long long)(uintptr_t)Whhb;

  // per-wave base for B-fragment reads from the stage buffer:
  // local row = w*64 + q*16 + c, byte = row*64 + half*16; q contributes q*1024 (imm)
  const unsigned sb_off = (unsigned)((w * 64 + c) * 64 + half * 16);

  __syncthreads();

  int cur = 0;
  for (int t = 0; t < T_DIM; ++t) {
    v8f acc[12] = {};
    // prologue: DMA stage 0 (kt=0, gate=0) into buffer 0
    if (w == 0) {
      tdm_load_whh(stage_off0, whh_base);
      __builtin_amdgcn_s_wait_tensorcnt(0);
    }
    __syncthreads();

    int s = 0;                                   // stage index = kt*3 + gate
#pragma unroll 1
    for (int kt = 0; kt < H_DIM / 32; ++kt) {
      // A fragment (h rows, bf16) — reused across the 3 gate stages of this kt
      Frag a;
      const unsigned short* ap = &hb[cur][c][kt * 32 + half * 8];
      a.q[0] = *(const uint4*)ap;
      a.q[1] = *(const uint4*)(ap + 16);
#pragma unroll
      for (int gate = 0; gate < 3; ++gate) {
        const int ns = s + 1;
        if (w == 0 && ns < 48) {                 // kick DMA for next stage
          int nkt = ns / 3, ngate = ns - nkt * 3;
          tdm_load_whh(stage_off0 + (unsigned)((ns & 1) * 32768),
                       whh_base + (unsigned long long)(ngate * 524288 + nkt * 64));
        }
        const unsigned char* sb = &stage[s & 1][0] + sb_off;
#pragma unroll
        for (int q = 0; q < 4; ++q) {
          Frag b;
          b.q[0] = *(const uint4*)(sb + q * 1024);
          b.q[1] = *(const uint4*)(sb + q * 1024 + 32);
          acc[gate * 4 + q] = __builtin_amdgcn_wmma_f32_16x16x32_bf16(
              false, a.v, false, b.v, (short)0, acc[gate * 4 + q], false, false);
        }
        if (w == 0 && ns < 48) __builtin_amdgcn_s_wait_tensorcnt(0);
        __syncthreads();                         // next stage buffer ready for everyone
        s = ns;
      }
    }

    const int nxt = cur ^ 1;
#pragma unroll
    for (int q = 0; q < 4; ++q) {
      int col = w * 64 + q * 16 + c;
#pragma unroll
      for (int v = 0; v < 8; ++v) {
        int m = half * 8 + v;
        size_t gxo = ((size_t)(rb + m) * T_DIM + t) * G_DIM;
        float iv = acc[q][v]     + bf2f(Gx[gxo + col]);
        float gv = acc[4 + q][v] + bf2f(Gx[gxo + H_DIM + col]);
        float ov = acc[8 + q][v] + bf2f(Gx[gxo + 2 * H_DIM + col]);
        float ct = sigmoidf_(iv) * tanhf(gv);    // forget-gate term vanishes (ref bug)
        float hv = sigmoidf_(ov) * tanhf(ct);
        hb[nxt][m][col] = f2bf(hv);
        if (t == T_DIM - 1) out[(size_t)(rb + m) * H_DIM + col] = hv;
      }
    }
    __syncthreads();                             // h(t+1) visible before next step
    cur = nxt;
  }
}

// ---------------- Launch ----------------
extern "C" void kernel_launch(void* const* d_in, const int* in_sizes, int n_in,
                              void* d_out, int out_size, void* d_ws, size_t ws_size,
                              hipStream_t stream) {
  const long long* inp  = (const long long*)d_in[0];  // [B,T] int64
  const float*     emb  = (const float*)d_in[1];      // [V,E]
  const float*     Wih  = (const float*)d_in[2];      // [4H,E]
  const float*     Whh  = (const float*)d_in[3];      // [4H,H]
  const float*     bih  = (const float*)d_in[4];      // [4H]
  const float*     bhh  = (const float*)d_in[5];      // [4H]
  float* out = (float*)d_out;                          // [B,H]

  char* ws = (char*)d_ws;
  unsigned short* Wihb  = (unsigned short*)(ws);                 // 1536*320*2 = 983040
  unsigned short* Whhb  = (unsigned short*)(ws + 983040);        // 1536*512*2 = 1572864
  float*          biasb = (float*)(ws + 2555904);                // 1536*4     = 6144
  unsigned short* Gx    = (unsigned short*)(ws + 2562048);       // 32768*1536*2 ~= 100.7 MB

  int prep_n = G_DIM * E_PAD + G_DIM * H_DIM + G_DIM;
  lstm_prep<<<(prep_n + 255) / 256, 256, 0, stream>>>(Wih, Whh, bih, bhh, Wihb, Whhb, biasb);
  lstm_xproj<<<(B_DIM * T_DIM) / 16, 256, 0, stream>>>(inp, emb, Wihb, biasb, Gx);
  lstm_recur<<<B_DIM / 16, 256, 0, stream>>>(Whhb, Gx, out);
}
